// RWKV_TimeMix_66434554134685
// MI455X (gfx1250) — matrix operand order repaired and checked by reading, not compiled, not run on hardware
//
#include <hip/hip_runtime.h>
#include <cstdint>
#include <cstddef>

// ---------------------------------------------------------------------------
// RWKV TimeMix for MI455X (gfx1250, wave32, WMMA)
// B=16, T=1024, C_EMBD=C_ATT=2048  -> M = B*T = 16384, N = K = 2048
// GEMMs run on v_wmma_f32_16x16x32_f16 (f16 in / f32 accumulate): per the
// per-instruction WMMA throughput table this is 8x the FLOP/instr of the
// f32 16x16x4 flavor, turning ~1.7ms of f32-matrix time into ~220us, while
// all elementwise math, accumulation and the WKV scan stay fp32.
// Pipeline:
//   1) k  = (x time-mixed with time_mix_k) @ key_w^T          [fused mix+GEMM]
//   2) v  = (x time-mixed with time_mix_v) @ value_w^T        [fused mix+GEMM]
//   3) sr = sigmoid((x mixed with time_mix_r) @ receptance_w^T) -> d_out (scratch)
//   4) WKV scan over T per (b,c) channel; y = sr * wkv  (written over k buffer)
//   5) out = y @ output_w^T                                   [GEMM]
// ---------------------------------------------------------------------------

#define B_DIM 16
#define T_DIM 1024
#define C_DIM 2048
#define M_DIM (B_DIM * T_DIM) // 16384

typedef _Float16 v4h  __attribute__((ext_vector_type(4)));
typedef _Float16 v8h  __attribute__((ext_vector_type(8)));
typedef _Float16 v16h __attribute__((ext_vector_type(16)));
typedef float    v8f  __attribute__((ext_vector_type(8)));

#define BM 128
#define BN 128
#define BK 32   // one 16x16x32 WMMA K-step per staged tile
#define LDST 40 // LDS row stride in halves: 80B rows -> 16B aligned, and the
                // 16 row-starts hit distinct 4-bank groups -> conflict-free b128

// GEMM: Out[m][n] = sum_c A'(m,c) * W[n][c]   (both operands K-contiguous)
//   MIX: A'(m,c) = mix[c]*x[m][c] + (1-mix[c])*x[m-1][c], with x[t=-1] = 0
//   SIG: apply sigmoid to the result (for receptance)
template <bool MIX, bool SIG>
__global__ __launch_bounds__(256)
void rwkv_gemm_kernel(const float* __restrict__ A,
                      const float* __restrict__ mix,
                      const float* __restrict__ W,
                      float* __restrict__ Out,
                      int M, int N, int K)
{
    __shared__ _Float16 Asl[2][BM * LDST];
    __shared__ _Float16 Bsl[2][BN * LDST];

    const int tid  = threadIdx.x;
    const int wave = tid >> 5;
    const int lane = tid & 31;
    const int l    = lane & 15;  // row/col-within-16 selector
    const int half = lane >> 4;  // K-chunk selector

    const int m0 = blockIdx.x * BM;
    const int n0 = blockIdx.y * BN;

    // 8 waves: 4 along M (32 rows each) x 2 along N (64 cols each)
    const int wm = (wave & 3) * 32;
    const int wn = (wave >> 2) * 64;

    v8f acc[2][4];
#pragma unroll
    for (int i = 0; i < 2; ++i)
#pragma unroll
        for (int j = 0; j < 4; ++j)
            acc[i][j] = v8f{};

    // Stage one BM x BK (and BN x BK) tile into LDS buffer `bs`, converting
    // f32 -> f16 and (for A) applying the fused time-shift mix.
    auto stage = [&](int bs, int k0) {
#pragma unroll
        for (int it = 0; it < 4; ++it) {
            const int linear = tid + it * 256;   // 0..1023
            const int row    = linear >> 3;      // 8 float4 per 32-float row
            const int col4   = (linear & 7) << 2;
            const int c      = k0 + col4;

            // A tile (optionally fused time-shift mix), f32 -> f16
            {
                const int m = m0 + row;
                const float4 xv = *(const float4*)(A + (size_t)m * K + c);
                float4 av = xv;
                if (MIX) {
                    const int t = m & (T_DIM - 1); // BM | T: tile stays in batch
                    float4 xp = make_float4(0.f, 0.f, 0.f, 0.f);
                    if (t != 0)
                        xp = *(const float4*)(A + (size_t)(m - 1) * K + c);
                    const float4 mx = *(const float4*)(mix + c);
                    av.x = fmaf(mx.x, xv.x - xp.x, xp.x);
                    av.y = fmaf(mx.y, xv.y - xp.y, xp.y);
                    av.z = fmaf(mx.z, xv.z - xp.z, xp.z);
                    av.w = fmaf(mx.w, xv.w - xp.w, xp.w);
                }
                v4h h = { (_Float16)av.x, (_Float16)av.y,
                          (_Float16)av.z, (_Float16)av.w };
                *(v4h*)&Asl[bs][row * LDST + col4] = h;
            }
            // B tile (weight rows, K-contiguous), f32 -> f16
            {
                const int n = n0 + row;
                const float4 wv = *(const float4*)(W + (size_t)n * K + c);
                v4h h = { (_Float16)wv.x, (_Float16)wv.y,
                          (_Float16)wv.z, (_Float16)wv.w };
                *(v4h*)&Bsl[bs][row * LDST + col4] = h;
            }
        }
    };

    stage(0, 0);
    __syncthreads();

    int bs = 0;
#pragma unroll 2
    for (int k0 = 0; k0 < K; k0 += BK) {
        // stage the next tile into the other buffer while computing this one
        if (k0 + BK < K)
            stage(bs ^ 1, k0 + BK);

        // ---- fragments for one 16x16x32 K-step ----
        // A 16x32 f16 layout: lane half h holds K in [8h,8h+8) and [16+8h,16+8h+8)
        v16h afr[2];
#pragma unroll
        for (int i = 0; i < 2; ++i) {
            const _Float16* p = &Asl[bs][(wm + i * 16 + l) * LDST + 8 * half];
            v8h lo = *(const v8h*)p;         // K = 8h .. 8h+7
            v8h hi = *(const v8h*)(p + 16);  // K = 16+8h .. 16+8h+7
            afr[i] = __builtin_shufflevector(lo, hi,
                0, 1, 2, 3, 4, 5, 6, 7, 8, 9, 10, 11, 12, 13, 14, 15);
        }
        // B 32x16: lane = column N, half h holds K in [8h,8h+8) and [16+8h,16+8h+8)
        v16h bfr[4];
#pragma unroll
        for (int j = 0; j < 4; ++j) {
            const _Float16* p = &Bsl[bs][(wn + j * 16 + l) * LDST + 8 * half];
            v8h lo = *(const v8h*)p;
            v8h hi = *(const v8h*)(p + 16);
            bfr[j] = __builtin_shufflevector(lo, hi,
                0, 1, 2, 3, 4, 5, 6, 7, 8, 9, 10, 11, 12, 13, 14, 15);
        }

#pragma unroll
        for (int i = 0; i < 2; ++i)
#pragma unroll
            for (int j = 0; j < 4; ++j)
                acc[i][j] = __builtin_amdgcn_wmma_f32_16x16x32_f16(
                    false, afr[i], false, bfr[j], (short)0, acc[i][j],
                    false, false);

        __syncthreads();
        bs ^= 1;
    }

    // ---- epilogue: C/D f32 layout: VGPR r -> row (r + 8*half), col = l ----
#pragma unroll
    for (int i = 0; i < 2; ++i) {
#pragma unroll
        for (int j = 0; j < 4; ++j) {
            const int ng = n0 + wn + j * 16 + l;
#pragma unroll
            for (int r = 0; r < 8; ++r) {
                const int mg = m0 + wm + i * 16 + half * 8 + r;
                float val = acc[i][j][r];
                if (SIG) val = 1.0f / (1.0f + __expf(-val));
                Out[(size_t)mg * N + ng] = val;
            }
        }
    }
}

// WKV recurrence: one thread per (b,c) channel, sequential over T, all fp32.
// Consecutive threads -> consecutive c -> fully coalesced loads/stores.
// ybuf may alias kbuf (each element is read at step t before being written).
__global__ __launch_bounds__(256)
void rwkv_wkv_kernel(const float* __restrict__ time_decay,
                     const float* __restrict__ time_first,
                     const float* kbuf,
                     const float* __restrict__ vbuf,
                     const float* __restrict__ srbuf,
                     float* ybuf)
{
    const int g = blockIdx.x * 256 + threadIdx.x; // 0 .. B*C-1
    const int c = g & (C_DIM - 1);
    const size_t base = (size_t)(g >> 11) * (size_t)T_DIM * C_DIM + c;

    const float w = -__expf(time_decay[c]);
    const float u = time_first[c];
    float aa = 0.f, bb = 0.f, pp = -1e38f;

    for (int t = 0; t < T_DIM; ++t) {
        const size_t idx = base + (size_t)t * C_DIM;
        const float kt = __builtin_nontemporal_load(kbuf + idx);  // read once
        const float vt = __builtin_nontemporal_load(vbuf + idx);  // read once
        const float sr = __builtin_nontemporal_load(srbuf + idx); // read once

        const float ww = u + kt;
        const float p  = fmaxf(pp, ww);
        const float e1 = __expf(pp - p);
        const float e2 = __expf(ww - p);
        // regular-temporal store: y (134MB) fits in the 192MB L2 for GEMM #5
        ybuf[idx] = sr * ((e1 * aa + e2 * vt) / (e1 * bb + e2));

        const float ww2 = pp + w;
        const float p2  = fmaxf(ww2, kt);
        const float f1  = __expf(ww2 - p2);
        const float f2  = __expf(kt - p2);
        aa = f1 * aa + f2 * vt;
        bb = f1 * bb + f2;
        pp = p2;
    }
}

extern "C" void kernel_launch(void* const* d_in, const int* in_sizes, int n_in,
                              void* d_out, int out_size, void* d_ws, size_t ws_size,
                              hipStream_t stream)
{
    (void)in_sizes; (void)n_in; (void)out_size; (void)ws_size;
    const float* x  = (const float*)d_in[0];
    const float* td = (const float*)d_in[1];
    const float* tf = (const float*)d_in[2];
    const float* mk = (const float*)d_in[3];
    const float* mv = (const float*)d_in[4];
    const float* mr = (const float*)d_in[5];
    const float* kw = (const float*)d_in[6];
    const float* vw = (const float*)d_in[7];
    const float* rw = (const float*)d_in[8];
    const float* ow = (const float*)d_in[9];
    float* out = (float*)d_out;

    const size_t elems = (size_t)M_DIM * C_DIM; // 33.5M floats = 134 MB
    float* kbuf  = (float*)d_ws;        // ws[0 .. 134MB): k, later rwkv (in-place)
    float* vbuf  = kbuf + elems;        // ws[134 .. 268MB): v
    float* srbuf = out;                 // d_out doubles as scratch for sigmoid(r)

    dim3 grid(M_DIM / BM, C_DIM / BN); // (128, 16)
    dim3 block(256);

    rwkv_gemm_kernel<true,  false><<<grid, block, 0, stream>>>(x, mk, kw, kbuf,  M_DIM, C_DIM, C_DIM);
    rwkv_gemm_kernel<true,  false><<<grid, block, 0, stream>>>(x, mv, vw, vbuf,  M_DIM, C_DIM, C_DIM);
    rwkv_gemm_kernel<true,  true ><<<grid, block, 0, stream>>>(x, mr, rw, srbuf, M_DIM, C_DIM, C_DIM);

    rwkv_wkv_kernel<<<dim3((B_DIM * C_DIM) / 256), block, 0, stream>>>(
        td, tf, kbuf, vbuf, srbuf, /*y in-place over k*/ kbuf);

    rwkv_gemm_kernel<false, false><<<grid, block, 0, stream>>>(kbuf, nullptr, ow, out, M_DIM, C_DIM, C_DIM);
}
